// WindowAttention_62680752718177
// MI455X (gfx1250) — matrix operand order repaired and checked by reading, not compiled, not run on hardware
//
#include <hip/hip_runtime.h>
#include <hip/hip_bf16.h>

// ---------------------------------------------------------------------------
// Swin window attention, fused, bf16 WMMA (gfx1250 / CDNA5, wave32)
// one workgroup (256 thr = 8 waves) per window of 49 tokens
// - prep kernel pre-converts weights to transposed bf16 in d_ws (if it fits)
// - weight staging via GLOBAL_LOAD_ASYNC_TO_LDS_B128 when available
// ---------------------------------------------------------------------------

typedef __bf16 bf16_t;
typedef __attribute__((ext_vector_type(16))) __bf16 v16bf;
typedef __attribute__((ext_vector_type(8)))  float  v8f;
typedef int v4i __attribute__((vector_size(16)));                 // b128 payload
typedef __attribute__((address_space(1))) v4i* gv4i_p;            // global
typedef __attribute__((address_space(3))) v4i* lv4i_p;            // LDS

#define NHEAD   8
#define NTOK    49      // tokens per window
#define CHAN    256
#define HDIM    32
#define QK_SCALE 0.17677669529663687f   // 32^-0.5

// LDS row strides (in bf16 elements), padded to dodge 64-bank conflicts
#define LX 264          // s_x   : 64 x 256  (x tile, later ctx tile)
#define LQ 776          // s_qkv : 64 x 768  (q|k|v, q pre-scaled)
#define LW 264          // s_w   : 128 x 256 (transposed weight chunk [n][k])
#define LP 72           // s_p   : 8 heads x 64 x 64 probs (aliases s_w)
#define LV 72           // s_vT  : 256 x 64  (v transposed: [d][token])

#define SQ_OFF (64 * LX)
#define SW_OFF (SQ_OFF + 64 * LQ)
#define SV_OFF (SW_OFF + NHEAD * 64 * LP)       // p-alias (36864) > w-chunk (33792)
#define SMEM_HALFS (SV_OFF + 256 * LV)          // ~238 KB total

#if defined(__HIP_DEVICE_COMPILE__) && __has_builtin(__builtin_amdgcn_global_load_async_to_lds_b128)
#define ASYNC_OK 1
#else
#define ASYNC_OK 0
#endif

// ---- WMMA wrapper ---------------------------------------------------------
__device__ __forceinline__ v8f wmma_bf16(v16bf a, v16bf b, v8f c) {
  // (neg_a, A, neg_b, B, c_mod, C, reuse_a, reuse_b)
  return __builtin_amdgcn_wmma_f32_16x16x32_bf16(false, a, false, b,
                                                 (short)0, c, false, false);
}

// ---- tile loaders (wave32), contiguous k order shared by A and B --------------------
// A 16x32 bf16: lane holds row = lane&15; elem e -> K=(lane>>4)*16+e.
//               src row-major [m][k].
__device__ __forceinline__ v16bf load_a(const bf16_t* src, int ld, int lane) {
  const bf16_t* p = src + (lane & 15) * ld + ((lane >> 4) * 16);
  v16bf a;
#pragma unroll
  for (int e = 0; e < 16; ++e) a[e] = p[e];
  return a;
}

// B 32x16 bf16 from a TRANSPOSED source (row-major [n][k]):
// lane holds col n = lane&15; elem e -> K = (lane>>4)*16 + e.  Contiguous.
__device__ __forceinline__ v16bf load_bT(const bf16_t* src, int ld, int lane) {
  const bf16_t* p = src + (lane & 15) * ld + ((lane >> 4) * 16);
  v16bf b;
#pragma unroll
  for (int e = 0; e < 16; ++e) b[e] = p[e];
  return b;
}

__device__ __forceinline__ v8f zero8() {
  v8f z;
#pragma unroll
  for (int e = 0; e < 8; ++e) z[e] = 0.0f;
  return z;
}

// ---- weight staging: async-to-LDS from bf16T workspace, or fp32 fallback --
// s_w[n][k] (n in [0,128), k in [0,256)), row stride LW.
__device__ __forceinline__ void stage_weights(bf16_t* __restrict__ s_w,
                                              const float*  __restrict__ wsrc,   // [256][n_total] fp32
                                              const bf16_t* __restrict__ wsrcT,  // [n_total][256] bf16
                                              int n_total, int col0,
                                              int use_ws, int tid)
{
  if (use_ws) {
#if ASYNC_OK
    // async DMA rows: each lane moves one 16B chunk, 16 rows per thread
    const int lo = (tid & 31) * 8;                       // halfs within row
    for (int row = tid >> 5; row < 128; row += 8) {
      const bf16_t* g = wsrcT + (size_t)(col0 + row) * 256 + lo;
      bf16_t* l = s_w + row * LW + lo;
      __builtin_amdgcn_global_load_async_to_lds_b128(
          (gv4i_p)g, (lv4i_p)l, 0, 0);
    }
#else
    for (int idx = tid; idx < 128 * 256; idx += 256) {
      int row = idx >> 8, k = idx & 255;
      s_w[row * LW + k] = wsrcT[(size_t)(col0 + row) * 256 + k];
    }
#endif
  } else {
    for (int idx = tid; idx < 256 * 128; idx += 256) {
      int k = idx >> 7, c = idx & 127;
      s_w[c * LW + k] = (bf16_t)wsrc[(size_t)k * n_total + col0 + c];
    }
  }
  // speculative prefetch of the next chunk (OOB is silently dropped)
  const char* nx = use_ws ? (const char*)(wsrcT + (size_t)(col0 + 128) * 256)
                          : (const char*)(wsrc + col0 + 128);
  __builtin_prefetch(nx + tid * 256, 0, 2);
}

__device__ __forceinline__ void wait_async_copies() {
#if ASYNC_OK
#if __has_builtin(__builtin_amdgcn_s_wait_asynccnt)
  __builtin_amdgcn_s_wait_asynccnt(0);
#else
  asm volatile("s_wait_asynccnt 0" ::: "memory");
#endif
#endif
}

// ---------------------------------------------------------------------------
// prep: w_qkv [256][768] -> ws[0 .. 768*256)      = wqkvT [n][k] bf16
//       w_proj [256][256] -> ws[768*256 .. +256*256) = wprojT [n][k] bf16
__global__ void __launch_bounds__(256)
convert_weights_kernel(const float* __restrict__ wqkv,
                       const float* __restrict__ wproj,
                       bf16_t* __restrict__ ws)
{
  int i = blockIdx.x * 256 + threadIdx.x;
  if (i < 768 * 256) {
    int n = i >> 8, k = i & 255;
    ws[i] = (bf16_t)wqkv[k * 768 + n];
  } else {
    int j = i - 768 * 256;
    int n = j >> 8, k = j & 255;
    ws[i] = (bf16_t)wproj[k * 256 + n];
  }
}

// ---------------------------------------------------------------------------
__global__ void __launch_bounds__(256)
swin_window_attn_kernel(const float* __restrict__ x,
                        const float* __restrict__ mask,
                        const int*   __restrict__ rel_index,
                        const float* __restrict__ w_qkv,
                        const float* __restrict__ b_qkv,
                        const float* __restrict__ bias_table,
                        const float* __restrict__ w_proj,
                        const float* __restrict__ b_proj,
                        const bf16_t* __restrict__ wsT,
                        int use_ws,
                        float* __restrict__ out)
{
  extern __shared__ bf16_t smem[];
  bf16_t* s_x   = smem;             // 64 x 256 (pad LX)
  bf16_t* s_qkv = smem + SQ_OFF;    // 64 x 768 (pad LQ)
  bf16_t* s_w   = smem + SW_OFF;    // 128 x 256 transposed weight chunk
  bf16_t* s_p   = s_w;              // alias: 8 x 64 x 64 probs
  bf16_t* s_vT  = smem + SV_OFF;    // 256 x 64 v transposed [d][token]

  const int tid    = threadIdx.x;
  const int lane   = tid & 31;
  const int wave   = tid >> 5;
  const int lanelo = lane & 15;
  const int half   = lane >> 4;
  const int blk    = blockIdx.x;
  const int winid  = blk & 63;      // mask window index = b % 64

  // per-wave output tile assignment for the dense GEMMs (M=64, N=128 chunks)
  const int mt  = wave >> 1;        // 0..3 : 16-row tile
  const int nt0 = (wave & 1) * 4;   // 0 or 4 : first of four 16-col tiles

  // ---- phase 0: stage x tile (fp32 -> bf16), zero-pad rows 49..63 ---------
  for (int idx = tid; idx < 64 * CHAN; idx += 256) {
    int m = idx >> 8, c = idx & 255;
    float v = (m < NTOK) ? x[((size_t)blk * NTOK + m) * CHAN + c] : 0.0f;
    s_x[m * LX + c] = (bf16_t)v;
  }

  // ---- phase 1: QKV GEMM  [64,256] x [256,768] in six 128-col chunks ------
  for (int chunk = 0; chunk < 6; ++chunk) {
    stage_weights(s_w, w_qkv, wsT, 768, chunk * 128, use_ws, tid);
    wait_async_copies();
    __syncthreads();

    v8f acc[4];
#pragma unroll
    for (int j = 0; j < 4; ++j) acc[j] = zero8();

    for (int kt = 0; kt < 8; ++kt) {
      v16bf a = load_a(s_x + mt * 16 * LX + kt * 32, LX, lane);
#pragma unroll
      for (int j = 0; j < 4; ++j) {
        v16bf bm = load_bT(s_w + (nt0 + j) * 16 * LW + kt * 32, LW, lane);
        acc[j] = wmma_bf16(a, bm, acc[j]);
      }
    }

    // epilogue: +bias, pre-scale q, store bf16 into s_qkv (+ transposed v)
#pragma unroll
    for (int j = 0; j < 4; ++j) {
#pragma unroll
      for (int r = 0; r < 8; ++r) {
        int m   = mt * 16 + r + 8 * half;
        int col = chunk * 128 + (nt0 + j) * 16 + lanelo;   // 0..767
        float v = acc[j][r] + b_qkv[col];
        if (col < 256) v *= QK_SCALE;                      // q columns
        s_qkv[m * LQ + col] = (bf16_t)v;
        if (col >= 512)                                    // v, also store d-major
          s_vT[(col - 512) * LV + m] = (bf16_t)v;
      }
    }
    __syncthreads();
  }

  // ---- phase 2: attention, one head per wave ------------------------------
  {
    const int h = wave;
    const bf16_t* qbase = s_qkv + h * HDIM;
    const bf16_t* kbase = s_qkv + 256 + h * HDIM;

    // scores S = q kT : 64x64 padded, K = 32 (one WMMA per tile)
    v8f sc[4][4];
    v16bf kb[4];
#pragma unroll
    for (int nt = 0; nt < 4; ++nt) {
      kb[nt] = load_bT(kbase + nt * 16 * LQ, LQ, lane);    // kT tile
#pragma unroll
      for (int m2 = 0; m2 < 4; ++m2) sc[m2][nt] = zero8();
    }
#pragma unroll
    for (int m2 = 0; m2 < 4; ++m2) {
      v16bf a = load_a(qbase + m2 * 16 * LQ, LQ, lane);
#pragma unroll
      for (int nt = 0; nt < 4; ++nt)
        sc[m2][nt] = wmma_bf16(a, kb[nt], sc[m2][nt]);
    }

    // +rel-pos bias, +window mask, column masking, register softmax
#pragma unroll
    for (int m2 = 0; m2 < 4; ++m2) {
#pragma unroll
      for (int r = 0; r < 8; ++r) {
        int m = m2 * 16 + r + 8 * half;
#pragma unroll
        for (int nt = 0; nt < 4; ++nt) {
          int n = nt * 16 + lanelo;
          float s = sc[m2][nt][r];
          if (n < NTOK) {
            if (m < NTOK) {
              int ri = rel_index[m * NTOK + n];
              s += bias_table[ri * NHEAD + h]
                 + mask[((size_t)winid * NTOK + m) * NTOK + n];
            }
          } else {
            s = -1e30f;                    // padded key columns
          }
          sc[m2][nt][r] = s;
        }
        // row-wise max over 4 regs + 16 lanes of this half-wave
        float mx = fmaxf(fmaxf(sc[m2][0][r], sc[m2][1][r]),
                         fmaxf(sc[m2][2][r], sc[m2][3][r]));
#pragma unroll
        for (int off = 1; off < 16; off <<= 1)
          mx = fmaxf(mx, __shfl_xor(mx, off, 32));
        float sum = 0.0f;
#pragma unroll
        for (int nt = 0; nt < 4; ++nt) {
          float p = __expf(sc[m2][nt][r] - mx);
          sc[m2][nt][r] = p;
          sum += p;
        }
#pragma unroll
        for (int off = 1; off < 16; off <<= 1)
          sum += __shfl_xor(sum, off, 32);
        float inv = 1.0f / sum;
#pragma unroll
        for (int nt = 0; nt < 4; ++nt)
          s_p[(h * 64 + m) * LP + nt * 16 + lanelo] =
              (bf16_t)(sc[m2][nt][r] * inv);
      }
    }
    __syncthreads();

    // ctx = P @ v : [64,64] x [64,32], K in two 32-steps; B from s_vT (contig)
    v8f ctx[4][2];
#pragma unroll
    for (int m2 = 0; m2 < 4; ++m2)
#pragma unroll
      for (int nt = 0; nt < 2; ++nt) ctx[m2][nt] = zero8();

#pragma unroll
    for (int kt = 0; kt < 2; ++kt) {
      v16bf vb[2];
#pragma unroll
      for (int nt = 0; nt < 2; ++nt)
        vb[nt] = load_bT(s_vT + (h * HDIM + nt * 16) * LV + kt * 32, LV, lane);
#pragma unroll
      for (int m2 = 0; m2 < 4; ++m2) {
        v16bf a = load_a(s_p + (h * 64 + m2 * 16) * LP + kt * 32, LP, lane);
#pragma unroll
        for (int nt = 0; nt < 2; ++nt)
          ctx[m2][nt] = wmma_bf16(a, vb[nt], ctx[m2][nt]);
      }
    }

    // scatter ctx (head-interleaved) back into s_x as bf16
#pragma unroll
    for (int m2 = 0; m2 < 4; ++m2)
#pragma unroll
      for (int nt = 0; nt < 2; ++nt)
#pragma unroll
        for (int r = 0; r < 8; ++r) {
          int m = m2 * 16 + r + 8 * half;
          s_x[m * LX + h * HDIM + nt * 16 + lanelo] = (bf16_t)ctx[m2][nt][r];
        }
    __syncthreads();
  }

  // ---- phase 3: projection GEMM [64,256] x [256,256] + bias -> global -----
  for (int chunk = 0; chunk < 2; ++chunk) {
    stage_weights(s_w, w_proj, wsT + 768 * 256, 256, chunk * 128, use_ws, tid);
    wait_async_copies();
    __syncthreads();

    v8f acc[4];
#pragma unroll
    for (int j = 0; j < 4; ++j) acc[j] = zero8();

    for (int kt = 0; kt < 8; ++kt) {
      v16bf a = load_a(s_x + mt * 16 * LX + kt * 32, LX, lane);
#pragma unroll
      for (int j = 0; j < 4; ++j) {
        v16bf bm = load_bT(s_w + (nt0 + j) * 16 * LW + kt * 32, LW, lane);
        acc[j] = wmma_bf16(a, bm, acc[j]);
      }
    }

#pragma unroll
    for (int j = 0; j < 4; ++j) {
#pragma unroll
      for (int r = 0; r < 8; ++r) {
        int m   = mt * 16 + r + 8 * half;
        int col = chunk * 128 + (nt0 + j) * 16 + lanelo;
        if (m < NTOK)
          out[((size_t)blk * NTOK + m) * CHAN + col] = acc[j][r] + b_proj[col];
      }
    }
    __syncthreads();
  }
}

// ---------------------------------------------------------------------------
extern "C" void kernel_launch(void* const* d_in, const int* in_sizes, int n_in,
                              void* d_out, int out_size, void* d_ws, size_t ws_size,
                              hipStream_t stream) {
  const float* x          = (const float*)d_in[0];
  const float* mask       = (const float*)d_in[1];
  const int*   rel_index  = (const int*)  d_in[2];
  const float* w_qkv      = (const float*)d_in[3];
  const float* b_qkv      = (const float*)d_in[4];
  const float* bias_table = (const float*)d_in[5];
  const float* w_proj     = (const float*)d_in[6];
  const float* b_proj     = (const float*)d_in[7];
  float* out = (float*)d_out;

  const int b_total = in_sizes[0] / (NTOK * CHAN);   // 4096 windows

  // pre-converted transposed bf16 weights in workspace, if it fits
  const size_t need = (size_t)(768 * 256 + 256 * 256) * sizeof(bf16_t);  // 512 KB
  bf16_t* wsT = (bf16_t*)d_ws;
  int use_ws = (d_ws != nullptr && ws_size >= need) ? 1 : 0;
  if (use_ws) {
    hipLaunchKernelGGL(convert_weights_kernel, dim3(1024), dim3(256), 0, stream,
                       w_qkv, w_proj, wsT);
  }

  const size_t smem_bytes = (size_t)SMEM_HALFS * sizeof(bf16_t);  // ~238 KB

  hipLaunchKernelGGL(swin_window_attn_kernel,
                     dim3(b_total), dim3(256), smem_bytes, stream,
                     x, mask, rel_index, w_qkv, b_qkv, bias_table,
                     w_proj, b_proj, wsT, use_ws, out);
}
